// GraphAttentionPooling_42803644072644
// MI455X (gfx1250) — matrix operand order repaired and checked by reading, not compiled, not run on hardware
//
#include <hip/hip_runtime.h>
#include <hip/hip_bf16.h>

// ---------------------------------------------------------------------------
// Problem constants (fixed by reference setup_inputs)
// ---------------------------------------------------------------------------
#define NN 4096     // nodes
#define EE 512      // embed dim
#define SS 64       // shards
#define HH 8        // heads
#define HD 64       // head dim
#define TOPK 15

// CDNA5 async global->LDS copy path (ASYNCcnt). Set to 0 to fall back to
// register-staged b128 copies.
#define USE_ASYNC_LDS_COPY 1

typedef __bf16 bf16_t;
typedef bf16_t bf16x16 __attribute__((ext_vector_type(16)));
typedef float  floatx8 __attribute__((ext_vector_type(8)));

union FragU {
    uint4   u[2];
    bf16x16 f;
};

// Per-lane async copy of 16 bytes global -> LDS. VDST holds the LDS byte
// address, VADDR the 64-bit global address (GV mode, SADDR=off).
__device__ __forceinline__ void async_copy_b128(unsigned int lds_addr, const void* gaddr)
{
#if USE_ASYNC_LDS_COPY
    asm volatile("global_load_async_to_lds_b128 %0, %1, off"
                 :: "v"(lds_addr), "v"(gaddr)
                 : "memory");
#endif
}

__device__ __forceinline__ void wait_async_copies()
{
#if USE_ASYNC_LDS_COPY
    asm volatile("s_wait_asynccnt 0x0" ::: "memory");
#endif
}

__device__ __forceinline__ unsigned int lds_byte_addr(const void* p)
{
    // generic LDS pointer: low 32 bits are the LDS byte offset
    return (unsigned int)(unsigned long long)p;
}

// ---------------------------------------------------------------------------
// GEMM:  C[m][n] = scale * sum_k A[m][k] * B[n][k]  + bias[n]     (B^T form)
//   A: bf16 row-major [M x K] (lda), B: bf16 row-major [N x K] (ldb)
//   Outputs: Cf (f32) and/or Cb (bf16), leading dim ldc. Either may be null.
// Workgroup: 256 threads = 8 waves. Block tile 128(M) x 64(N), BK=32.
//   waves arranged 2(M) x 4(N); each wave owns 64x16 = 4 WMMA subtiles.
// Double-buffered LDS staging via async global->LDS b128 copies: tile t+1
// streams in while tile t runs on the WMMA pipe; one barrier per K-step.
// Requires: M%128==0, N%64==0, K%32==0; all tile addresses 16-B aligned.
// ---------------------------------------------------------------------------
__global__ __launch_bounds__(256) void gemm_bf16_wmma(
    const __hip_bfloat16* __restrict__ A, int lda,
    const __hip_bfloat16* __restrict__ B, int ldb,
    const float* __restrict__ bias, float scale,
    float* __restrict__ Cf, __hip_bfloat16* __restrict__ Cb, int ldc,
    int M, int N, int K)
{
    // rows padded to 40 halves (80 B = 16*5) so 16-byte ds accesses stay aligned
    __shared__ __align__(16) unsigned short As[2][128 * 40];
    __shared__ __align__(16) unsigned short Bs[2][64 * 40];

    const int tid    = threadIdx.x;
    const int lane   = tid & 31;
    const int wid    = tid >> 5;
    const int wave_m = wid >> 2;   // 0..1
    const int wave_n = wid & 3;    // 0..3
    const int tile_m = blockIdx.y * 128;
    const int tile_n = blockIdx.x * 64;
    const int half   = lane >> 4;  // K-half selector per ISA fragment layout
    const int lr     = lane & 15;

    floatx8 acc[4];
    #pragma unroll
    for (int i = 0; i < 4; ++i)
        #pragma unroll
        for (int r = 0; r < 8; ++r) acc[i][r] = 0.0f;

    // issue the cooperative copy of one 128x32 A tile + 64x32 B tile
    auto issue_tile = [&](int k0, int buf) {
        // A: 512 x b128, 2 per thread
        #pragma unroll
        for (int i = 0; i < 2; ++i) {
            int u   = tid + i * 256;
            int row = u >> 2;
            int qp  = u & 3;
            const void* g = A + (size_t)(tile_m + row) * lda + k0 + qp * 8;
#if USE_ASYNC_LDS_COPY
            async_copy_b128(lds_byte_addr(&As[buf][row * 40 + qp * 8]), g);
#else
            *reinterpret_cast<uint4*>(&As[buf][row * 40 + qp * 8]) =
                *reinterpret_cast<const uint4*>(g);
#endif
        }
        // B: 256 x b128, 1 per thread
        {
            int n  = tid >> 2;
            int qp = tid & 3;
            const void* g = B + (size_t)(tile_n + n) * ldb + k0 + qp * 8;
#if USE_ASYNC_LDS_COPY
            async_copy_b128(lds_byte_addr(&Bs[buf][n * 40 + qp * 8]), g);
#else
            *reinterpret_cast<uint4*>(&Bs[buf][n * 40 + qp * 8]) =
                *reinterpret_cast<const uint4*>(g);
#endif
        }
    };

    const int T = K >> 5;          // K-steps of 32
    issue_tile(0, 0);              // prologue: stage first tile

    for (int t = 0; t < T; ++t) {
        const int cur = t & 1;
        wait_async_copies();       // this wave's copies for tile t done
        __syncthreads();           // all waves' tile-t writes visible;
                                   // all reads of buffer cur^1 (tile t-1) done
        if (t + 1 < T)
            issue_tile((t + 1) << 5, cur ^ 1);   // overlap with compute below

        // ---- B fragment: elem j <-> B[k = half*16 + j][n], n = wave_n*16+lr ----
        FragU fb;
        {
            const unsigned short* bp = &Bs[cur][(wave_n * 16 + lr) * 40 + half * 16];
            fb.u[0] = *reinterpret_cast<const uint4*>(bp);
            fb.u[1] = *reinterpret_cast<const uint4*>(bp + 8);
        }
        // ---- 4 A fragments + WMMA ----
        #pragma unroll
        for (int i = 0; i < 4; ++i) {
            const int m = wave_m * 64 + i * 16 + lr;
            FragU fa;
            // elems 0..15 <-> k = half*16 .. half*16+15
            fa.u[0] = *reinterpret_cast<const uint4*>(&As[cur][m * 40 + half * 16]);
            fa.u[1] = *reinterpret_cast<const uint4*>(&As[cur][m * 40 + half * 16 + 8]);
            acc[i] = __builtin_amdgcn_wmma_f32_16x16x32_bf16(
                false, fa.f, false, fb.f, (short)0, acc[i], false, false);
        }
    }

    // ---- epilogue: D vgpr r, lanes 0-15 -> M=r, lanes 16-31 -> M=r+8; N=lr ----
    const int n_g = tile_n + wave_n * 16 + lr;
    const float bv = bias ? bias[n_g] : 0.0f;
    #pragma unroll
    for (int i = 0; i < 4; ++i) {
        const int m_base = tile_m + wave_m * 64 + i * 16 + half * 8;
        #pragma unroll
        for (int r = 0; r < 8; ++r) {
            float v = acc[i][r] * scale + bv;
            size_t off = (size_t)(m_base + r) * ldc + n_g;
            if (Cf) Cf[off] = v;
            if (Cb) Cb[off] = __float2bfloat16(v);
        }
    }
}

// ---------------------------------------------------------------------------
// f32 -> bf16 convert
// ---------------------------------------------------------------------------
__global__ void cvt_bf16_kernel(const float* __restrict__ in,
                                __hip_bfloat16* __restrict__ out, int n)
{
    for (int i = blockIdx.x * blockDim.x + threadIdx.x; i < n;
         i += gridDim.x * blockDim.x)
        out[i] = __float2bfloat16(in[i]);
}

__global__ void fill_zero_kernel(float* __restrict__ p, int n)
{
    for (int i = blockIdx.x * blockDim.x + threadIdx.x; i < n;
         i += gridDim.x * blockDim.x)
        p[i] = 0.0f;
}

// v section of qkv [4096 x 512] -> vT [512 x 4096] so the PV GEMM can use the
// B^T fast path (all-async b128 tile fills). 8 MB traffic, L2-resident.
__global__ void transpose_v_kernel(const __hip_bfloat16* __restrict__ qkvb,
                                   __hip_bfloat16* __restrict__ vT)
{
    int idx = blockIdx.x * 256 + threadIdx.x;   // over EE*NN
    int m = idx & (NN - 1);
    int e = idx >> 12;
    vT[(size_t)e * NN + m] = qkvb[(size_t)m * (3 * EE) + 2 * EE + e];
}

// ---------------------------------------------------------------------------
// Row softmax on [4096 x 4096] scores; writes bf16 attn and accumulates
// 0.125 * attn into the head-mean output. One block (256 thr) per row.
// ---------------------------------------------------------------------------
__global__ __launch_bounds__(256) void softmax_attn_kernel(
    const float* __restrict__ scores, __hip_bfloat16* __restrict__ attnb,
    float* __restrict__ attn_mean)
{
    __shared__ float buf[NN];
    __shared__ float red[256];
    const int row = blockIdx.x;
    const int t = threadIdx.x;
    const float* srow = scores + (size_t)row * NN;

    float lmax = -__builtin_inff();
    for (int c = t; c < NN; c += 256) { float v = srow[c]; buf[c] = v; lmax = fmaxf(lmax, v); }
    red[t] = lmax; __syncthreads();
    for (int s = 128; s > 0; s >>= 1) {
        if (t < s) red[t] = fmaxf(red[t], red[t + s]);
        __syncthreads();
    }
    const float m = red[0]; __syncthreads();

    float lsum = 0.0f;
    for (int c = t; c < NN; c += 256) { float e = expf(buf[c] - m); buf[c] = e; lsum += e; }
    red[t] = lsum; __syncthreads();
    for (int s = 128; s > 0; s >>= 1) {
        if (t < s) red[t] += red[t + s];
        __syncthreads();
    }
    const float inv = 1.0f / red[0];

    for (int c = t; c < NN; c += 256) {
        float p = buf[c] * inv;
        attnb[(size_t)row * NN + c] = __float2bfloat16(p);
        attn_mean[(size_t)row * NN + c] += 0.125f * p;  // mean over 8 heads
    }
}

// ---------------------------------------------------------------------------
// Prototype MLP (tiny: 64 x 512, K=512) in plain fp32
// ---------------------------------------------------------------------------
__global__ void proto_mlp1_kernel(const float* __restrict__ enh,
                                  const float* __restrict__ w1,
                                  const float* __restrict__ b1,
                                  float* __restrict__ h)
{
    int idx = blockIdx.x * 256 + threadIdx.x;
    if (idx >= SS * EE) return;
    int s = idx >> 9, j = idx & 511;
    const float* x = enh + (size_t)(s * (NN / SS)) * EE;   // idx_s = s*64
    const float* w = w1 + (size_t)j * EE;
    float acc = b1[j];
    for (int e = 0; e < EE; ++e) acc += x[e] * w[e];
    h[idx] = fmaxf(acc, 0.0f);
}

__global__ void proto_mlp2_kernel(const float* __restrict__ h,
                                  const float* __restrict__ w2,
                                  const float* __restrict__ b2,
                                  float* __restrict__ protos)
{
    int idx = blockIdx.x * 256 + threadIdx.x;
    if (idx >= SS * EE) return;
    int s = idx >> 9, e = idx & 511;
    const float* hh = h + (size_t)s * EE;
    const float* w  = w2 + (size_t)e * EE;
    float acc = b2[e];
    for (int j = 0; j < EE; ++j) acc += hh[j] * w[j];
    protos[idx] = acc;
}

// L2-normalize 64 prototype rows in place. grid 64, block 128.
__global__ __launch_bounds__(128) void proto_norm_kernel(float* __restrict__ protos)
{
    __shared__ float red[128];
    const int n = blockIdx.x, t = threadIdx.x;
    float l = 0.0f;
    for (int e = t; e < EE; e += 128) { float v = protos[(size_t)n * EE + e]; l += v * v; }
    red[t] = l; __syncthreads();
    for (int s = 64; s > 0; s >>= 1) { if (t < s) red[t] += red[t + s]; __syncthreads(); }
    const float inv = 1.0f / fmaxf(sqrtf(red[0]), 1e-12f);
    __syncthreads();
    for (int e = t; e < EE; e += 128) protos[(size_t)n * EE + e] *= inv;
}

// row norms of enhanced. grid 4096, block 128.
__global__ __launch_bounds__(128) void enorm_kernel(const float* __restrict__ enh,
                                                    float* __restrict__ enorm)
{
    __shared__ float red[128];
    const int n = blockIdx.x, t = threadIdx.x;
    float l = 0.0f;
    for (int e = t; e < EE; e += 128) { float v = enh[(size_t)n * EE + e]; l += v * v; }
    red[t] = l; __syncthreads();
    for (int s = 64; s > 0; s >>= 1) { if (t < s) red[t] += red[t + s]; __syncthreads(); }
    if (t == 0) enorm[n] = fmaxf(sqrtf(red[0]), 1e-12f);
}

// similarity [4096 x 64]; grid 4096, block 64 (thread = shard)
__global__ __launch_bounds__(64) void sim_kernel(const float* __restrict__ enh,
                                                 const float* __restrict__ enorm,
                                                 const float* __restrict__ protosn,
                                                 float* __restrict__ sim)
{
    __shared__ float erow[EE];
    const int n = blockIdx.x, s = threadIdx.x;
    for (int e = s; e < EE; e += 64) erow[e] = enh[(size_t)n * EE + e];
    __syncthreads();
    const float* p = protosn + (size_t)s * EE;
    float acc = 0.0f;
    for (int e = 0; e < EE; ++e) acc += erow[e] * p[e];
    sim[(size_t)n * SS + s] = acc / enorm[n];
}

// ---------------------------------------------------------------------------
// std(similarity) -> adaptive temperature. stats[0]=sum, [1]=sumsq, [2]=temp, [3]=flag
// ---------------------------------------------------------------------------
__global__ void stats_init_kernel(float* stats)
{
    if (blockIdx.x == 0 && threadIdx.x < 4) stats[threadIdx.x] = 0.0f;
}

__global__ __launch_bounds__(256) void stats_reduce_kernel(const float* __restrict__ sim,
                                                           float* __restrict__ stats)
{
    __shared__ float rs[256], rss[256];
    const int t = threadIdx.x;
    float s = 0.0f, ss = 0.0f;
    for (int i = blockIdx.x * 256 + t; i < NN * SS; i += gridDim.x * 256) {
        float v = sim[i]; s += v; ss += v * v;
    }
    rs[t] = s; rss[t] = ss; __syncthreads();
    for (int st = 128; st > 0; st >>= 1) {
        if (t < st) { rs[t] += rs[t + st]; rss[t] += rss[t + st]; }
        __syncthreads();
    }
    if (t == 0) { atomicAdd(&stats[0], rs[0]); atomicAdd(&stats[1], rss[0]); }
}

__global__ void stats_final_kernel(float* stats)
{
    if (blockIdx.x == 0 && threadIdx.x == 0) {
        const float n = (float)(NN * SS);
        float mean = stats[0] / n;
        float var  = (stats[1] - n * mean * mean) / (n - 1.0f);  // unbiased
        float sd   = sqrtf(fmaxf(var, 0.0f));
        stats[2]   = 25.0f + sd * 10.0f;   // base_temp = max(1.0, 25.0)
    }
}

// softmax(sim / temp) per row. grid 4096, block 64.
__global__ __launch_bounds__(64) void assign_softmax_kernel(const float* __restrict__ sim,
                                                            const float* __restrict__ stats,
                                                            float* __restrict__ assign)
{
    __shared__ float red[64];
    const int n = blockIdx.x, s = threadIdx.x;
    const float x = sim[(size_t)n * SS + s] / stats[2];
    red[s] = x; __syncthreads();
    for (int st = 32; st > 0; st >>= 1) { if (s < st) red[s] = fmaxf(red[s], red[s + st]); __syncthreads(); }
    const float m = red[0]; __syncthreads();
    const float e = expf(x - m);
    red[s] = e; __syncthreads();
    for (int st = 32; st > 0; st >>= 1) { if (s < st) red[s] += red[s + st]; __syncthreads(); }
    assign[(size_t)n * SS + s] = e / red[0];
}

// column sums of assignment. mode=1: write (sum<51.2)?2:0 ; mode=0: write sum.
__global__ __launch_bounds__(256) void colsum_kernel(const float* __restrict__ a,
                                                     float* __restrict__ out, int mode)
{
    __shared__ float red[256];
    const int s = blockIdx.x, t = threadIdx.x;
    float l = 0.0f;
    for (int r = t; r < NN; r += 256) l += a[(size_t)r * SS + s];
    red[t] = l; __syncthreads();
    for (int st = 128; st > 0; st >>= 1) { if (t < st) red[t] += red[t + st]; __syncthreads(); }
    if (t == 0) {
        float sum = red[0];
        out[s] = mode ? ((sum < (float)(NN / SS) * 0.8f) ? 2.0f : 0.0f) : sum;
    }
}

// per-shard top-15 over column of similarity, scatter-add underval[s].
__global__ __launch_bounds__(256) void topk_add_kernel(const float* __restrict__ sim,
                                                       const float* __restrict__ underval,
                                                       float* __restrict__ assign)
{
    __shared__ float vals[NN];
    __shared__ float rv[256];
    __shared__ int   ri[256];
    __shared__ int   topIdx[TOPK];
    const int s = blockIdx.x, t = threadIdx.x;
    for (int r = t; r < NN; r += 256) vals[r] = sim[(size_t)r * SS + s];
    __syncthreads();
    for (int it = 0; it < TOPK; ++it) {
        float bv = -__builtin_inff();
        int   bi = 0x7fffffff;
        for (int r = t; r < NN; r += 256) {
            float v = vals[r];
            if (v > bv || (v == bv && r < bi)) { bv = v; bi = r; }
        }
        rv[t] = bv; ri[t] = bi; __syncthreads();
        for (int st = 128; st > 0; st >>= 1) {
            if (t < st) {
                float ov = rv[t + st]; int oi = ri[t + st];
                if (ov > rv[t] || (ov == rv[t] && oi < ri[t])) { rv[t] = ov; ri[t] = oi; }
            }
            __syncthreads();
        }
        if (t == 0) { topIdx[it] = ri[0]; vals[ri[0]] = -__builtin_inff(); }
        __syncthreads();
    }
    const float add = underval[s];
    if (t < TOPK) assign[(size_t)topIdx[t] * SS + s] += add;
}

// L1 renormalize rows. grid 4096, block 64.
__global__ __launch_bounds__(64) void renorm_rows_kernel(float* __restrict__ a)
{
    __shared__ float red[64];
    const int n = blockIdx.x, s = threadIdx.x;
    const float v = a[(size_t)n * SS + s];
    red[s] = fabsf(v); __syncthreads();
    for (int st = 32; st > 0; st >>= 1) { if (s < st) red[s] += red[s + st]; __syncthreads(); }
    a[(size_t)n * SS + s] = v / fmaxf(red[0], 1e-12f);
}

__global__ __launch_bounds__(64) void cond_kernel(const float* __restrict__ fp,
                                                  float* __restrict__ stats)
{
    __shared__ float mx[64], mn[64];
    const int t = threadIdx.x;
    mx[t] = fp[t]; mn[t] = fp[t]; __syncthreads();
    for (int st = 32; st > 0; st >>= 1) {
        if (t < st) { mx[t] = fmaxf(mx[t], mx[t + st]); mn[t] = fminf(mn[t], mn[t + st]); }
        __syncthreads();
    }
    if (t == 0) stats[3] = (mx[0] > mn[0] * 3.0f) ? 1.0f : 0.0f;
}

__global__ void blend_kernel(float* __restrict__ a, const float* __restrict__ stats, int n)
{
    int i = blockIdx.x * blockDim.x + threadIdx.x;
    if (i < n && stats[3] != 0.0f)
        a[i] = 0.3f * a[i] + 0.7f * (1.0f / (float)SS);
}

// ---------------------------------------------------------------------------
// Orchestration
// ---------------------------------------------------------------------------
extern "C" void kernel_launch(void* const* d_in, const int* in_sizes, int n_in,
                              void* d_out, int out_size, void* d_ws, size_t ws_size,
                              hipStream_t stream)
{
    (void)in_sizes; (void)n_in; (void)out_size; (void)ws_size;

    const float* emb       = (const float*)d_in[0];
    const float* in_proj_w = (const float*)d_in[2];
    const float* in_proj_b = (const float*)d_in[3];
    const float* out_w     = (const float*)d_in[4];
    const float* out_b     = (const float*)d_in[5];
    const float* w1        = (const float*)d_in[6];
    const float* b1        = (const float*)d_in[7];
    const float* w2        = (const float*)d_in[8];
    const float* b2        = (const float*)d_in[9];

    float* out_assign = (float*)d_out;                    // [4096 x 64]
    float* out_enh    = out_assign + (size_t)NN * SS;     // [4096 x 512]
    float* out_attn   = out_enh + (size_t)NN * EE;        // [4096 x 4096]

    // workspace carve-up
    char* wp = (char*)d_ws;
    auto carve = [&](size_t bytes) -> void* {
        void* p = (void*)wp;
        wp += (bytes + 255) & ~(size_t)255;
        return p;
    };
    __hip_bfloat16* xb     = (__hip_bfloat16*)carve((size_t)NN * EE * 2);       // embeddings bf16
    __hip_bfloat16* wqkvb  = (__hip_bfloat16*)carve((size_t)3 * EE * EE * 2);   // in_proj_w bf16
    __hip_bfloat16* outwb  = (__hip_bfloat16*)carve((size_t)EE * EE * 2);       // out_w bf16
    __hip_bfloat16* qkvb   = (__hip_bfloat16*)carve((size_t)NN * 3 * EE * 2);   // qkv bf16 [4096 x 1536]
    __hip_bfloat16* vT     = (__hip_bfloat16*)carve((size_t)EE * NN * 2);       // v transposed [512 x 4096]
    float*          scores = (float*)carve((size_t)NN * NN * 4);                // per-head scores f32
    __hip_bfloat16* attnb  = (__hip_bfloat16*)carve((size_t)NN * NN * 2);       // per-head attn bf16
    __hip_bfloat16* ctxb   = (__hip_bfloat16*)carve((size_t)NN * EE * 2);       // attn @ v, all heads
    float*          protoh = (float*)carve((size_t)SS * EE * 4);
    float*          protos = (float*)carve((size_t)SS * EE * 4);
    float*          enorm  = (float*)carve((size_t)NN * 4);
    float*          simbuf = (float*)carve((size_t)NN * SS * 4);
    float*          underv = (float*)carve((size_t)SS * 4);
    float*          fp     = (float*)carve((size_t)SS * 4);
    float*          stats  = (float*)carve(64);

    // ---- conversions ----
    cvt_bf16_kernel<<<(NN * EE) / 256, 256, 0, stream>>>(emb, xb, NN * EE);
    cvt_bf16_kernel<<<(3 * EE * EE) / 256, 256, 0, stream>>>(in_proj_w, wqkvb, 3 * EE * EE);
    cvt_bf16_kernel<<<(EE * EE) / 256, 256, 0, stream>>>(out_w, outwb, EE * EE);

    // zero the attention-mean output accumulator (deterministic across replays)
    fill_zero_kernel<<<(NN * NN) / 256, 256, 0, stream>>>(out_attn, NN * NN);

    // ---- QKV projection: qkv = x @ W^T + b   [4096 x 1536] bf16 ----
    gemm_bf16_wmma<<<dim3((3 * EE) / 64, NN / 128), 256, 0, stream>>>(
        xb, EE, wqkvb, EE, in_proj_b, 1.0f,
        nullptr, qkvb, 3 * EE, NN, 3 * EE, EE);

    // ---- v -> vT so PV GEMM uses the B^T (all-async) path ----
    transpose_v_kernel<<<(EE * NN) / 256, 256, 0, stream>>>(qkvb, vT);

    // ---- per-head attention ----
    for (int h = 0; h < HH; ++h) {
        // scores = (q_h @ k_h^T) / sqrt(hd)   [4096 x 4096] f32
        gemm_bf16_wmma<<<dim3(NN / 64, NN / 128), 256, 0, stream>>>(
            qkvb + h * HD, 3 * EE,            // A = q_h
            qkvb + EE + h * HD, 3 * EE,       // B = k_h (row n' = key n')
            nullptr, 0.125f,                  // 1/sqrt(64)
            scores, nullptr, NN, NN, NN, HD);
        // softmax rows -> bf16 attn, accumulate 1/8 into mean output
        softmax_attn_kernel<<<NN, 256, 0, stream>>>(scores, attnb, out_attn);
        // ctx_h = attn @ v_h   [4096 x 64] bf16 into ctxb columns [h*64, h*64+64)
        gemm_bf16_wmma<<<dim3(HD / 64, NN / 128), 256, 0, stream>>>(
            attnb, NN,
            vT + (size_t)h * HD * NN, NN,     // B = v_h^T [64 x 4096]
            nullptr, 1.0f,
            nullptr, ctxb + h * HD, EE, NN, HD, NN);
    }

    // ---- output projection: enhanced = ctx @ out_w^T + out_b  (f32 output) ----
    gemm_bf16_wmma<<<dim3(EE / 64, NN / 128), 256, 0, stream>>>(
        ctxb, EE, outwb, EE, out_b, 1.0f,
        out_enh, nullptr, EE, NN, EE, EE);

    // ---- prototype MLP (fp32, tiny) ----
    proto_mlp1_kernel<<<(SS * EE) / 256, 256, 0, stream>>>(out_enh, w1, b1, protoh);
    proto_mlp2_kernel<<<(SS * EE) / 256, 256, 0, stream>>>(protoh, w2, b2, protos);
    proto_norm_kernel<<<SS, 128, 0, stream>>>(protos);

    // ---- cosine similarity ----
    enorm_kernel<<<NN, 128, 0, stream>>>(out_enh, enorm);
    sim_kernel<<<NN, 64, 0, stream>>>(out_enh, enorm, protos, simbuf);

    // ---- adaptive temperature from unbiased std ----
    stats_init_kernel<<<1, 64, 0, stream>>>(stats);
    stats_reduce_kernel<<<256, 256, 0, stream>>>(simbuf, stats);
    stats_final_kernel<<<1, 64, 0, stream>>>(stats);

    // ---- assignment = softmax(sim / temp) ----
    assign_softmax_kernel<<<NN, 64, 0, stream>>>(simbuf, stats, out_assign);

    // ---- under-allocated shards + top-15 scatter-add ----
    colsum_kernel<<<SS, 256, 0, stream>>>(out_assign, underv, /*mode=*/1);
    topk_add_kernel<<<SS, 256, 0, stream>>>(simbuf, underv, out_assign);

    // ---- L1 renorm rows ----
    renorm_rows_kernel<<<NN, 64, 0, stream>>>(out_assign);

    // ---- conditional uniform blend ----
    colsum_kernel<<<SS, 256, 0, stream>>>(out_assign, fp, /*mode=*/0);
    cond_kernel<<<1, 64, 0, stream>>>(fp, stats);
    blend_kernel<<<(NN * SS) / 256, 256, 0, stream>>>(out_assign, stats, NN * SS);
}
